// GraRPINet_37924561223999
// MI455X (gfx1250) — compile-verified, matrix operand order checked by reading, not run on hardware
//
#include <hip/hip_runtime.h>
#include <hip/hip_bf16.h>

typedef __attribute__((ext_vector_type(16))) _Float16 v16h;
typedef __attribute__((ext_vector_type(8)))  float    v8f;

#define HEADS 4

// ---------------------------------------------------------------------------
// Pre-swizzle weights: fp32 W[K,Ncols] -> f16 fragments in exact WMMA B-layout,
// K zero-padded to a multiple of 32. One thread per (chunk, colTile, lane)
// writes the 16 contiguous halves that this lane will load in the GEMM.
// Fragment order: frag = (chunk * tiles + tile); lane block = 32 bytes.
// ---------------------------------------------------------------------------
__global__ void swizzle_w(const float* __restrict__ W, _Float16* __restrict__ out,
                          int K, int Ncols)
{
    const int KC = (K + 31) >> 5;
    const int tiles = Ncols >> 4;
    const int total = KC * tiles * 32;
    int i = blockIdx.x * blockDim.x + threadIdx.x;
    if (i >= total) return;
    const int lane = i & 31;
    const int tile = (i >> 5) % tiles;
    const int kc   = (i >> 5) / tiles;
    const int n    = lane & 15;
    const int kgrp = (lane >> 4) << 3;
    const int col  = tile * 16 + n;
    _Float16* op = out + (size_t)i * 16;
#pragma unroll
    for (int j = 0; j < 8; ++j) {
        int ka = kc * 32 + kgrp + j;
        int kb = ka + 16;
        op[j]     = (ka < K) ? (_Float16)W[(size_t)ka * Ncols + col] : (_Float16)0.0f;
        op[j + 8] = (kb < K) ? (_Float16)W[(size_t)kb * Ncols + col] : (_Float16)0.0f;
    }
}

// ---------------------------------------------------------------------------
// GEMM: C[M,Ncols] = A[M,K] @ W[K,Ncols] (+ bias), fp32 A/C, pre-swizzled f16 W.
// K compile-time -> branch-free; A fragment = 4x b128 loads + packed cvts;
// B fragment = single contiguous 32B v16h load per tile. NT=4 col tiles per
// wave share one A fragment (4 accumulators). 8 waves/block = 128 rows.
// All control flow wave-uniform -> EXEC all-1s at each WMMA (ISA req).
// ---------------------------------------------------------------------------
template<int K>
__global__ void wmma_gemm_f16(const float* __restrict__ A, const _Float16* __restrict__ swz,
                              const float* __restrict__ bias, float* __restrict__ C,
                              int M, int lda, int ldc, int coloff, int tiles)
{
    constexpr int NT = 4;
    constexpr int NCH = K >> 5;                 // full 32-K chunks
    constexpr bool TAIL = (K & 31) != 0;
    const int wave = threadIdx.x >> 5;
    const int lane = threadIdx.x & 31;
    const int row0 = (blockIdx.x * 8 + wave) << 4;
    const int tile0 = blockIdx.y * NT;
    if (row0 >= M) return;                      // wave-uniform exit

    const int n    = lane & 15;
    const int kgrp = (lane >> 4) << 3;

    const float* __restrict__ arow = A + (size_t)(row0 + n) * lda;
    const _Float16* __restrict__ wlane = swz + ((size_t)lane << 4);

    v8f acc[NT];
#pragma unroll
    for (int t = 0; t < NT; ++t) acc[t] = v8f{};

    for (int c = 0; c < NCH; ++c) {
        const int k0 = c << 5;
        // A 16x32 f16 fragment: lanes0-15 hold K=[0..7],[16..23]; lanes16-31 K=[8..15],[24..31]
        const float4 a0 = *(const float4*)(arow + k0 + kgrp);
        const float4 a1 = *(const float4*)(arow + k0 + kgrp + 4);
        const float4 a2 = *(const float4*)(arow + k0 + kgrp + 16);
        const float4 a3 = *(const float4*)(arow + k0 + kgrp + 20);
        v16h a;
        a[0]=(_Float16)a0.x; a[1]=(_Float16)a0.y; a[2]=(_Float16)a0.z; a[3]=(_Float16)a0.w;
        a[4]=(_Float16)a1.x; a[5]=(_Float16)a1.y; a[6]=(_Float16)a1.z; a[7]=(_Float16)a1.w;
        a[8]=(_Float16)a2.x; a[9]=(_Float16)a2.y; a[10]=(_Float16)a2.z; a[11]=(_Float16)a2.w;
        a[12]=(_Float16)a3.x; a[13]=(_Float16)a3.y; a[14]=(_Float16)a3.z; a[15]=(_Float16)a3.w;

#pragma unroll
        for (int t = 0; t < NT; ++t) {
            const v16h b = *(const v16h*)(wlane + (((size_t)c * tiles + tile0 + t) << 9));
            acc[t] = __builtin_amdgcn_wmma_f32_16x16x32_f16(
                false, a, false, b, (short)0, acc[t], false, false);
        }
    }

    if constexpr (TAIL) {                       // A masked (select-style); B pre-zero-padded
        v16h a;
#pragma unroll
        for (int i = 0; i < 8; ++i) {
            int ka = NCH * 32 + kgrp + i;
            int kb = ka + 16;
            bool pa = ka < K, pb = kb < K;
            float av0 = arow[pa ? ka : 0];
            float av1 = arow[pb ? kb : 0];
            a[i]     = (_Float16)(pa ? av0 : 0.0f);
            a[i + 8] = (_Float16)(pb ? av1 : 0.0f);
        }
#pragma unroll
        for (int t = 0; t < NT; ++t) {
            const v16h b = *(const v16h*)(wlane + (((size_t)NCH * tiles + tile0 + t) << 9));
            acc[t] = __builtin_amdgcn_wmma_f32_16x16x32_f16(
                false, a, false, b, (short)0, acc[t], false, false);
        }
    }

    // C/D f32 layout: VGPR r -> row r (lanes 0-15) / row r+8 (lanes 16-31), N = lane&15
    const int rbase = row0 + ((lane >> 4) << 3);
#pragma unroll
    for (int t = 0; t < NT; ++t) {
        const int col = (tile0 + t) * 16 + n;
        const float bv = bias ? bias[col] : 0.0f;
        float* __restrict__ cp = C + coloff + col;
#pragma unroll
        for (int r = 0; r < 8; ++r) {
            cp[(size_t)(rbase + r) * ldc] = acc[t][r] + bv;
        }
    }
}

// ---------------------------------------------------------------------------
// Small helper kernels (edge softmax / attention machinery)
// ---------------------------------------------------------------------------
__global__ void fill_f32(float* __restrict__ p, float v, int n)
{
    int i = blockIdx.x * blockDim.x + threadIdx.x;
    if (i < n) p[i] = v;
}

// fold att_d into Wd:  wd_att[k,h] = sum_c Wd[k, h*C+c] * att_d[h,c]
__global__ void make_wdatt(const float* __restrict__ Wd, const float* __restrict__ att_d,
                           float* __restrict__ wdatt, int K, int C)
{
    int i = blockIdx.x * blockDim.x + threadIdx.x;
    if (i >= K * HEADS) return;
    int k = i >> 2, h = i & 3;
    const float* wp = Wd + (size_t)k * HEADS * C + h * C;
    const float* ap = att_d + h * C;
    float s = 0.f;
    for (int c = 0; c < C; ++c) s += wp[c] * ap[c];
    wdatt[i] = s;
}

// a_s[n,h] = dot(xs[n,h,:], att_s[h,:])
__global__ void attn_dot(const float* __restrict__ xs, const float* __restrict__ att,
                         float* __restrict__ out, int Nn, int C)
{
    int i = blockIdx.x * blockDim.x + threadIdx.x;
    if (i >= Nn * HEADS) return;
    int n = i >> 2, h = i & 3;
    const float* xp = xs + (size_t)n * HEADS * C + h * C;
    const float* ap = att + h * C;
    float s = 0.f;
    for (int c = 0; c < C; ++c) s += xp[c] * ap[c];
    out[i] = s;
}

// a_d[n,h] = dot(h_in[n,:], wd_att[:,h])
__global__ void ad_dot(const float* __restrict__ hin, const float* __restrict__ wdatt,
                       float* __restrict__ out, int Nn, int K)
{
    int i = blockIdx.x * blockDim.x + threadIdx.x;
    if (i >= Nn * HEADS) return;
    int n = i >> 2, h = i & 3;
    const float* hp = hin + (size_t)n * K;
    float s = 0.f;
    for (int k = 0; k < K; ++k) s += hp[k] * wdatt[k * HEADS + h];
    out[i] = s;
}

__device__ __forceinline__ void atomicMaxFloat(float* addr, float val)
{
    if (!(val < 0.0f)) atomicMax((int*)addr, __float_as_int(val));
    else               atomicMin((unsigned int*)addr, __float_as_uint(val));
}

// pass 1: e = leaky_relu(a_s[src]+a_d[dst]); running max per (dst,head)
__global__ void edge_max(const int* __restrict__ src, const int* __restrict__ dst,
                         const float* __restrict__ a_s, const float* __restrict__ a_d,
                         float* __restrict__ ebuf, float* __restrict__ mmax, int E)
{
    int i = blockIdx.x * blockDim.x + threadIdx.x;
    if (i >= E * HEADS) return;
    int e = i >> 2, h = i & 3;
    int s = src[e], d = dst[e];
    float v = a_s[s * HEADS + h] + a_d[d * HEADS + h];
    v = (v > 0.f) ? v : 0.2f * v;                // leaky relu, slope 0.2
    ebuf[i] = v;
    atomicMaxFloat(&mmax[d * HEADS + h], v);
}

// pass 2: p = exp(e - m[dst]); denom[dst] += p   (in-place e -> p)
__global__ void edge_exp(const int* __restrict__ dst, float* __restrict__ pbuf,
                         const float* __restrict__ mmax, float* __restrict__ denom, int E)
{
    int i = blockIdx.x * blockDim.x + threadIdx.x;
    if (i >= E * HEADS) return;
    int e = i >> 2, h = i & 3;
    int d = dst[e];
    float p = expf(pbuf[i] - mmax[d * HEADS + h]);
    pbuf[i] = p;
    atomicAdd(&denom[d * HEADS + h], p);
}

// pass 3: agg[dst] += (p/denom[dst]) * xs[src]; one thread per (edge, 4 channels)
__global__ void edge_agg(const int* __restrict__ src, const int* __restrict__ dst,
                         const float* __restrict__ pbuf, const float* __restrict__ denom,
                         const float* __restrict__ xs, float* __restrict__ agg,
                         int E, int C)
{
    const int HC = HEADS * C;
    const int vecs = HC >> 2;                    // float4 groups per edge
    int i = blockIdx.x * blockDim.x + threadIdx.x;
    if (i >= E * vecs) return;
    int e = i / vecs, j4 = i % vecs;
    int h = (j4 << 2) / C;
    int s = src[e], d = dst[e];
    float alpha = pbuf[e * HEADS + h] / denom[d * HEADS + h];
    const float4 xv = ((const float4*)(xs + (size_t)s * HC))[j4];
    float* ag = agg + (size_t)d * HC + (j4 << 2);
    atomicAdd(ag + 0, alpha * xv.x);
    atomicAdd(ag + 1, alpha * xv.y);
    atomicAdd(ag + 2, alpha * xv.z);
    atomicAdd(ag + 3, alpha * xv.w);
}

// h1 = relu(lin + agg + bias)  (in place into lin)
__global__ void combine_relu(float* __restrict__ lin, const float* __restrict__ agg,
                             const float* __restrict__ bias, int total, int cols)
{
    int i = blockIdx.x * blockDim.x + threadIdx.x;
    if (i >= total) return;
    float v = lin[i] + agg[i] + bias[i % cols];
    lin[i] = (v > 0.f) ? v : 0.f;
}

// out += agg + bias  (out already holds lin2 result)
__global__ void final_add(float* __restrict__ out, const float* __restrict__ agg,
                          const float* __restrict__ bias, int total, int cols)
{
    int i = blockIdx.x * blockDim.x + threadIdx.x;
    if (i >= total) return;
    out[i] += agg[i] + bias[i % cols];
}

// ---------------------------------------------------------------------------
extern "C" void kernel_launch(void* const* d_in, const int* in_sizes, int n_in,
                              void* d_out, int out_size, void* d_ws, size_t ws_size,
                              hipStream_t stream)
{
    const float* x       = (const float*)d_in[0];   // [N,1300]
    const int*   eidx    = (const int*)  d_in[1];   // [2,E]
    const float* W_rna   = (const float*)d_in[3];   // [1000,64]
    const float* b_rna   = (const float*)d_in[4];
    const float* W_prot  = (const float*)d_in[5];   // [300,64]
    const float* b_prot  = (const float*)d_in[6];
    const float* W1s     = (const float*)d_in[7];   // [128,256]
    const float* W1d     = (const float*)d_in[8];
    const float* att1_s  = (const float*)d_in[9];   // [4,64]
    const float* att1_d  = (const float*)d_in[10];
    const float* bias1   = (const float*)d_in[11];  // [256]
    const float* Wl1     = (const float*)d_in[12];
    const float* bl1     = (const float*)d_in[13];
    const float* W2s     = (const float*)d_in[14];  // [256,128]
    const float* W2d     = (const float*)d_in[15];
    const float* att2_s  = (const float*)d_in[16];  // [4,32]
    const float* att2_d  = (const float*)d_in[17];
    const float* bias2   = (const float*)d_in[18];  // [128]
    const float* Wl2     = (const float*)d_in[19];
    const float* bl2     = (const float*)d_in[20];
    float* out = (float*)d_out;

    const int N = in_sizes[0] / 1300;
    const int E = in_sizes[1] / 2;
    const int* src = eidx;
    const int* dst = eidx + E;

    // workspace layout (floats, then f16 swizzled weights)
    float* ws = (float*)d_ws;
    float* h0   = ws;                               // [N,128]
    float* xs   = h0  + (size_t)N * 128;            // [N,256] (reused as [N,128] in L2)
    float* lin  = xs  + (size_t)N * 256;            // [N,256] -> becomes h1 in place
    float* agg  = lin + (size_t)N * 256;            // [N,256]
    float* a_s  = agg + (size_t)N * 256;            // [N,4]
    float* a_d  = a_s + (size_t)N * HEADS;          // [N,4]
    float* mx   = a_d + (size_t)N * HEADS;          // [N,4]
    float* den  = mx  + (size_t)N * HEADS;          // [N,4]
    float* pb   = den + (size_t)N * HEADS;          // [E,4]
    float* wd   = pb  + (size_t)E * HEADS;          // [256,4]
    _Float16* swz_rna  = (_Float16*)(wd + 1024);    // 32 chunks x 4 tiles x 512
    _Float16* swz_prot = swz_rna  + 32 * 4 * 512;   // 10 x 4 x 512
    _Float16* swz_w1s  = swz_prot + 10 * 4 * 512;   // 4 x 16 x 512
    _Float16* swz_wl1  = swz_w1s  +  4 * 16 * 512;
    _Float16* swz_w2s  = swz_wl1  +  4 * 16 * 512;  // 8 x 8 x 512
    _Float16* swz_wl2  = swz_w2s  +  8 * 8 * 512;

    const int T = 256;
    const int gemm_rb = (N + 127) / 128;
    auto cdiv = [](int a, int b) { return (a + b - 1) / b; };

    // ---- pre-swizzle all weights into WMMA B-fragment layout (f16, K padded)
    swizzle_w<<<cdiv(32 * 4 * 32, T), T, 0, stream>>>(W_rna,  swz_rna,  1000, 64);
    swizzle_w<<<cdiv(10 * 4 * 32, T), T, 0, stream>>>(W_prot, swz_prot,  300, 64);
    swizzle_w<<<cdiv(4 * 16 * 32, T), T, 0, stream>>>(W1s,    swz_w1s,   128, 256);
    swizzle_w<<<cdiv(4 * 16 * 32, T), T, 0, stream>>>(Wl1,    swz_wl1,   128, 256);
    swizzle_w<<<cdiv(8 *  8 * 32, T), T, 0, stream>>>(W2s,    swz_w2s,   256, 128);
    swizzle_w<<<cdiv(8 *  8 * 32, T), T, 0, stream>>>(Wl2,    swz_wl2,   256, 128);

    // ---- embedding: h0[:, :64] = x[:, :1000]@W_rna + b ; h0[:, 64:] = x[:,1000:]@W_prot + b
    wmma_gemm_f16<1000><<<dim3(gemm_rb, 1), T, 0, stream>>>(x,        swz_rna,  b_rna,  h0, N, 1300, 128, 0,  4);
    wmma_gemm_f16< 300><<<dim3(gemm_rb, 1), T, 0, stream>>>(x + 1000, swz_prot, b_prot, h0, N, 1300, 128, 64, 4);

    // ---- layer 1 projections
    wmma_gemm_f16<128><<<dim3(gemm_rb, 4), T, 0, stream>>>(h0, swz_w1s, nullptr, xs,  N, 128, 256, 0, 16);
    wmma_gemm_f16<128><<<dim3(gemm_rb, 4), T, 0, stream>>>(h0, swz_wl1, bl1,     lin, N, 128, 256, 0, 16);

    make_wdatt<<<cdiv(128 * HEADS, T), T, 0, stream>>>(W1d, att1_d, wd, 128, 64);
    attn_dot  <<<cdiv(N * HEADS, T),   T, 0, stream>>>(xs, att1_s, a_s, N, 64);
    ad_dot    <<<cdiv(N * HEADS, T),   T, 0, stream>>>(h0, wd, a_d, N, 128);

    // ---- layer 1 segment softmax + aggregation
    fill_f32<<<cdiv(N * HEADS, T), T, 0, stream>>>(mx, -1e30f, N * HEADS);
    fill_f32<<<cdiv(N * HEADS, T), T, 0, stream>>>(den, 0.0f,  N * HEADS);
    fill_f32<<<cdiv(N * 256, T),   T, 0, stream>>>(agg, 0.0f,  N * 256);
    edge_max<<<cdiv(E * HEADS, T), T, 0, stream>>>(src, dst, a_s, a_d, pb, mx, E);
    edge_exp<<<cdiv(E * HEADS, T), T, 0, stream>>>(dst, pb, mx, den, E);
    edge_agg<<<cdiv(E * 64, T),    T, 0, stream>>>(src, dst, pb, den, xs, agg, E, 64);

    // ---- combine + relu  (lin becomes h1 in place)
    combine_relu<<<cdiv(N * 256, T), T, 0, stream>>>(lin, agg, bias1, N * 256, 256);
    float* h1 = lin;

    // ---- layer 2 projections (lin2 written directly to d_out)
    wmma_gemm_f16<256><<<dim3(gemm_rb, 2), T, 0, stream>>>(h1, swz_w2s, nullptr, xs,  N, 256, 128, 0, 8);
    wmma_gemm_f16<256><<<dim3(gemm_rb, 2), T, 0, stream>>>(h1, swz_wl2, bl2,     out, N, 256, 128, 0, 8);

    make_wdatt<<<cdiv(256 * HEADS, T), T, 0, stream>>>(W2d, att2_d, wd, 256, 32);
    attn_dot  <<<cdiv(N * HEADS, T),   T, 0, stream>>>(xs, att2_s, a_s, N, 32);
    ad_dot    <<<cdiv(N * HEADS, T),   T, 0, stream>>>(h1, wd, a_d, N, 256);

    // ---- layer 2 segment softmax + aggregation
    fill_f32<<<cdiv(N * HEADS, T), T, 0, stream>>>(mx, -1e30f, N * HEADS);
    fill_f32<<<cdiv(N * HEADS, T), T, 0, stream>>>(den, 0.0f,  N * HEADS);
    fill_f32<<<cdiv(N * 128, T),   T, 0, stream>>>(agg, 0.0f,  N * 128);
    edge_max<<<cdiv(E * HEADS, T), T, 0, stream>>>(src, dst, a_s, a_d, pb, mx, E);
    edge_exp<<<cdiv(E * HEADS, T), T, 0, stream>>>(dst, pb, mx, den, E);
    edge_agg<<<cdiv(E * 32, T),    T, 0, stream>>>(src, dst, pb, den, xs, agg, E, 32);

    // ---- final: out = lin2 + agg + bias2
    final_add<<<cdiv(N * 128, T), T, 0, stream>>>(out, agg, bias2, N * 128, 128);
}
